// LSTMLMHeadModel_21423296872454
// MI455X (gfx1250) — matrix operand order repaired and checked by reading, not compiled, not run on hardware
//
#include <hip/hip_runtime.h>
#include <hip/hip_bf16.h>
#include <math.h>

typedef __attribute__((ext_vector_type(16))) __bf16 v16bf;
typedef __attribute__((ext_vector_type(8)))  __bf16 v8bf;
typedef __attribute__((ext_vector_type(8)))  float  v8f;

// ---- problem constants (from reference) ----
#define BATCH 4
#define SEQ   1024
#define EMB   1024
#define HID   1296            // int(1024*1.266)
#define HIDP  1344            // HID padded to multiple of 64 (even # of 32-wide K chunks)
#define GATES (4*HID)         // 5184 (multiple of 64)
#define MROWS (BATCH*SEQ)     // 4096
#define VOC   32003
#define VOCP  32064           // padded to multiple of 64 (4-wide N register blocking)

__device__ __forceinline__ __bf16 f2bf(float f) {
    unsigned u = __builtin_bit_cast(unsigned, f);
    u += 0x7FFFu + ((u >> 16) & 1u);            // round-to-nearest-even
    unsigned short h = (unsigned short)(u >> 16);
    return __builtin_bit_cast(__bf16, h);
}

__device__ __forceinline__ float sigm(float x) { return 1.0f / (1.0f + __expf(-x)); }

// ------------------------------------------------------------------
// C[M,N] = A[M,K](bf16, row-major, lda) * B[N,K]^T (bf16, row-major, ldb)
// One wave per (16*MT)x(16*NT) register-blocked macro tile.
// Software-pipelined: two fragment register sets ping-pong so chunk k+1's
// loads are in flight while chunk k's WMMAs issue. kChunks must be even, >=2.
// Stores only columns n < nValid; adds bias[n] if non-null.
// ------------------------------------------------------------------
template<int MT, int NT>
__global__ void gemm_bf16_wmma(const __bf16* __restrict__ A, int lda,
                               const __bf16* __restrict__ Bw, int ldb,
                               float* __restrict__ C, size_t ldc,
                               const float* __restrict__ bias,
                               int nValid, int kChunks)
{
    const int lane = threadIdx.x;
    const int l    = lane & 15;
    const int half = lane >> 4;
    const int m0   = blockIdx.y * (16 * MT);
    const int n0   = blockIdx.x * (16 * NT);

    const __bf16* arow[MT];
    const __bf16* brow[NT];
#pragma unroll
    for (int mt = 0; mt < MT; ++mt)
        arow[mt] = A + (size_t)(m0 + mt * 16 + l) * (size_t)lda;
#pragma unroll
    for (int nt = 0; nt < NT; ++nt)
        brow[nt] = Bw + (size_t)(n0 + nt * 16 + l) * (size_t)ldb;

    v8f acc[MT][NT] = {};

    // A 16x32 bf16 layout: lanes 0-15 hold K {0..7,16..23}, lanes 16-31 K {8..15,24..31}
    auto loadA = [&](int k0, v16bf (&a)[MT]) {
#pragma unroll
        for (int mt = 0; mt < MT; ++mt) {
            v8bf lo = *reinterpret_cast<const v8bf*>(arow[mt] + k0 + half * 8);
            v8bf hi = *reinterpret_cast<const v8bf*>(arow[mt] + k0 + 16 + half * 8);
#pragma unroll
            for (int i = 0; i < 8; ++i) { a[mt][i] = lo[i]; a[mt][i + 8] = hi[i]; }
        }
    };
    // B 32x16 bf16 layout: lane = column; lanes 0-15 hold K {0..7,16..23}, lanes 16-31 K {8..15,24..31}
    auto loadB = [&](int k0, v16bf (&b)[NT]) {
#pragma unroll
        for (int nt = 0; nt < NT; ++nt) {
            v8bf lo = *reinterpret_cast<const v8bf*>(brow[nt] + k0 + half * 8);
            v8bf hi = *reinterpret_cast<const v8bf*>(brow[nt] + k0 + 16 + half * 8);
#pragma unroll
            for (int i = 0; i < 8; ++i) { b[nt][i] = lo[i]; b[nt][i + 8] = hi[i]; }
        }
    };
    auto mma = [&](v16bf (&a)[MT], v16bf (&b)[NT]) {
#pragma unroll
        for (int mt = 0; mt < MT; ++mt)
#pragma unroll
            for (int nt = 0; nt < NT; ++nt)
                acc[mt][nt] = __builtin_amdgcn_wmma_f32_16x16x32_bf16(
                    false, a[mt], false, b[nt], (short)0, acc[mt][nt], false, false);
    };

    v16bf a0[MT], b0[NT], a1[MT], b1[NT];
    loadA(0, a0); loadB(0, b0);

    for (int kc = 0; kc < kChunks - 2; kc += 2) {
        loadA((kc + 1) * 32, a1); loadB((kc + 1) * 32, b1);
#pragma unroll
        for (int nt = 0; nt < NT; ++nt)   // 1 KiB (16 chunks) ahead of the 2-chunk consume rate
            __builtin_prefetch((const void*)(brow[nt] + kc * 32 + 512), 0, 1);
        mma(a0, b0);
        loadA((kc + 2) * 32, a0); loadB((kc + 2) * 32, b0);
        mma(a1, b1);
    }
    loadA((kChunks - 1) * 32, a1); loadB((kChunks - 1) * 32, b1);
    mma(a0, b0);
    mma(a1, b1);

#pragma unroll
    for (int nt = 0; nt < NT; ++nt) {
        const int n = n0 + nt * 16 + l;
        if (n < nValid) {
            const float bv = bias ? bias[n] : 0.0f;
#pragma unroll
            for (int mt = 0; mt < MT; ++mt)
#pragma unroll
                for (int r = 0; r < 8; ++r) {
                    const int m = m0 + mt * 16 + half * 8 + r;  // C/D layout: half selects M block of 8
                    C[(size_t)m * ldc + n] = acc[mt][nt][r] + bv;
                }
        }
    }
}

// ---- zero fill (u32 granularity; every buffer is 4B-multiple) ----
__global__ void zero_u32(unsigned* __restrict__ p, size_t n) {
    size_t i = (size_t)blockIdx.x * blockDim.x + threadIdx.x;
    if (i < n) p[i] = 0u;
}

// ---- f32 -> bf16 with zero padding: dst is [Rd x Cd], src is [Rs x Cs] ----
__global__ void cvt_f32_bf16_pad(const float* __restrict__ W, __bf16* __restrict__ Wb,
                                 int Cs, int Cd, int Rs, size_t total) {
    size_t i = (size_t)blockIdx.x * blockDim.x + threadIdx.x;
    if (i >= total) return;
    int r = (int)(i / (size_t)Cd);
    int c = (int)(i % (size_t)Cd);
    float v = (r < Rs && c < Cs) ? W[(size_t)r * Cs + c] : 0.0f;
    Wb[i] = f2bf(v);
}

// ---- embedding gather -> bf16 activations [MROWS, EMB] ----
__global__ void embed_bf16(const int* __restrict__ ids, const float* __restrict__ emb,
                           __bf16* __restrict__ x) {
    size_t i = (size_t)blockIdx.x * blockDim.x + threadIdx.x;
    if (i >= (size_t)MROWS * EMB) return;
    int row = (int)(i / EMB);
    int e   = (int)(i % EMB);
    x[i] = f2bf(emb[(size_t)ids[row] * EMB + e]);
}

// ---- LSTM pointwise step: gates -> (c,h); writes h as bf16 for next step + y sequence ----
__global__ void lstm_step(const float* __restrict__ gbuf,   // [16, GATES] (rows 0..3 valid)
                          const float* __restrict__ xg,     // [MROWS, GATES]
                          const float* __restrict__ b_ih,
                          const float* __restrict__ b_hh,
                          float* __restrict__ c,            // [BATCH, HID]
                          __bf16* __restrict__ hbf,         // [16, HIDP] (padding stays 0)
                          __bf16* __restrict__ ybf,         // [MROWS, HIDP]
                          int t)
{
    int idx = blockIdx.x * blockDim.x + threadIdx.x;
    if (idx >= BATCH * HID) return;
    int b = idx / HID, j = idx % HID;
    size_t grow = (size_t)b * GATES;
    size_t xrow = (size_t)(b * SEQ + t) * GATES;

    float gi = gbuf[grow + j]         + xg[xrow + j]         + b_ih[j]         + b_hh[j];
    float gf = gbuf[grow + HID + j]   + xg[xrow + HID + j]   + b_ih[HID + j]   + b_hh[HID + j];
    float gg = gbuf[grow + 2*HID + j] + xg[xrow + 2*HID + j] + b_ih[2*HID + j] + b_hh[2*HID + j];
    float go = gbuf[grow + 3*HID + j] + xg[xrow + 3*HID + j] + b_ih[3*HID + j] + b_hh[3*HID + j];

    float iv = sigm(gi);
    float fv = sigm(gf);
    float gv = tanhf(gg);
    float ov = sigm(go);

    float cn = fv * c[idx] + iv * gv;
    c[idx] = cn;
    float h = ov * tanhf(cn);

    __bf16 hb = f2bf(h);
    hbf[(size_t)b * HIDP + j] = hb;
    ybf[(size_t)(b * SEQ + t) * HIDP + j] = hb;
}

// ---- per-row log-softmax NLL, masked accumulate ----
__global__ void loss_rows(const float* __restrict__ logits,  // [MROWS, VOC]
                          const int* __restrict__ labels,    // [BATCH, SEQ]
                          float* __restrict__ accum)         // {sum_nll, sum_mask}
{
    int row = blockIdx.x;                 // 0 .. BATCH*(SEQ-1)-1
    int b = row / (SEQ - 1);
    int s = row % (SEQ - 1);
    const float* lg = logits + (size_t)(b * SEQ + s) * VOC;

    __shared__ float red[256];
    float mx = -3.4e38f;
    for (int j = threadIdx.x; j < VOC; j += blockDim.x) mx = fmaxf(mx, lg[j]);
    red[threadIdx.x] = mx; __syncthreads();
    for (int st = 128; st > 0; st >>= 1) {
        if ((int)threadIdx.x < st) red[threadIdx.x] = fmaxf(red[threadIdx.x], red[threadIdx.x + st]);
        __syncthreads();
    }
    mx = red[0]; __syncthreads();

    float sum = 0.0f;
    for (int j = threadIdx.x; j < VOC; j += blockDim.x) sum += __expf(lg[j] - mx);
    red[threadIdx.x] = sum; __syncthreads();
    for (int st = 128; st > 0; st >>= 1) {
        if ((int)threadIdx.x < st) red[threadIdx.x] += red[threadIdx.x + st];
        __syncthreads();
    }

    if (threadIdx.x == 0) {
        int lab = labels[b * SEQ + s + 1];
        if (lab != 0) {  // PAD == 0
            float lse = mx + __logf(red[0]);
            atomicAdd(&accum[0], lse - lg[lab]);
            atomicAdd(&accum[1], 1.0f);
        }
    }
}

__global__ void loss_finalize(const float* __restrict__ accum, float* __restrict__ out_loss) {
    if (threadIdx.x == 0 && blockIdx.x == 0)
        out_loss[0] = accum[0] / fmaxf(accum[1], 1.0f);
}

// ------------------------------------------------------------------
extern "C" void kernel_launch(void* const* d_in, const int* in_sizes, int n_in,
                              void* d_out, int out_size, void* d_ws, size_t ws_size,
                              hipStream_t stream) {
    (void)in_sizes; (void)n_in; (void)out_size; (void)ws_size;

    const int*   input_ids = (const int*)  d_in[0];
    const int*   labels    = (const int*)  d_in[1];
    const float* emb       = (const float*)d_in[2];
    const float* W_ih0     = (const float*)d_in[3];
    const float* W_hh0     = (const float*)d_in[4];
    const float* b_ih0     = (const float*)d_in[5];
    const float* b_hh0     = (const float*)d_in[6];
    const float* W_ih1     = (const float*)d_in[7];
    const float* W_hh1     = (const float*)d_in[8];
    const float* b_ih1     = (const float*)d_in[9];
    const float* b_hh1     = (const float*)d_in[10];
    const float* W_out     = (const float*)d_in[11];
    const float* b_out     = (const float*)d_in[12];

    float* logits = (float*)d_out;                        // [MROWS, VOC]
    float* loss   = logits + (size_t)MROWS * VOC;         // scalar

    // ---- workspace carve-up (256B aligned) ----
    char* ws = (char*)d_ws;
    size_t off = 0;
    auto carve = [&](size_t bytes) -> char* {
        char* p = ws + off;
        off = (off + bytes + 255) & ~(size_t)255;
        return p;
    };
    __bf16* wih0_bf = (__bf16*)carve((size_t)GATES * EMB  * 2);
    __bf16* whh0_bf = (__bf16*)carve((size_t)GATES * HIDP * 2);
    __bf16* wih1_bf = (__bf16*)carve((size_t)GATES * HIDP * 2);
    __bf16* whh1_bf = (__bf16*)carve((size_t)GATES * HIDP * 2);
    __bf16* wout_bf = (__bf16*)carve((size_t)VOCP  * HIDP * 2);
    __bf16* x0_bf   = (__bf16*)carve((size_t)MROWS * EMB  * 2);
    __bf16* y0_bf   = (__bf16*)carve((size_t)MROWS * HIDP * 2);
    __bf16* y1_bf   = (__bf16*)carve((size_t)MROWS * HIDP * 2);
    float*  xg      = (float*) carve((size_t)MROWS * GATES * 4);
    float*  gbuf    = (float*) carve((size_t)16 * GATES * 4);
    __bf16* hbf     = (__bf16*)carve((size_t)16 * HIDP * 2);
    float*  cbuf    = (float*) carve((size_t)BATCH * HID * 4);
    float*  accum   = (float*) carve(2 * 4);

    auto blocks = [](size_t n, int t) { return (unsigned)((n + t - 1) / t); };

    // ---- weight/activation precision conversion (bf16, zero-padded) ----
    {
        size_t t;
        t = (size_t)GATES * EMB;
        cvt_f32_bf16_pad<<<blocks(t,256),256,0,stream>>>(W_ih0, wih0_bf, EMB,  EMB,  GATES, t);
        t = (size_t)GATES * HIDP;
        cvt_f32_bf16_pad<<<blocks(t,256),256,0,stream>>>(W_hh0, whh0_bf, HID,  HIDP, GATES, t);
        cvt_f32_bf16_pad<<<blocks(t,256),256,0,stream>>>(W_ih1, wih1_bf, HID,  HIDP, GATES, t);
        cvt_f32_bf16_pad<<<blocks(t,256),256,0,stream>>>(W_hh1, whh1_bf, HID,  HIDP, GATES, t);
        t = (size_t)VOCP * HIDP;
        cvt_f32_bf16_pad<<<blocks(t,256),256,0,stream>>>(W_out, wout_bf, HID,  HIDP, VOC,  t);
        t = (size_t)MROWS * EMB;
        embed_bf16<<<blocks(t,256),256,0,stream>>>(input_ids, emb, x0_bf);
        // zero y buffers so padded columns stay 0 for downstream K-reads
        t = (size_t)MROWS * HIDP * 2 / 4;
        zero_u32<<<blocks(t,256),256,0,stream>>>((unsigned*)y0_bf, t);
        zero_u32<<<blocks(t,256),256,0,stream>>>((unsigned*)y1_bf, t);
    }

    const dim3 wave(32, 1, 1);
    const dim3 gridXG(GATES / 64, MROWS / 32);   // 81 x 128, <2,4> macro tiles
    const dim3 gridREC(GATES / 16, 1);           // 324 x 1,  <1,1> (M=16, latency-bound)
    const dim3 gridLOG(VOCP / 64, MROWS / 32);   // 501 x 128, <2,4>

    // =================== layer 0 ===================
    gemm_bf16_wmma<2,4><<<gridXG, wave, 0, stream>>>(x0_bf, EMB, wih0_bf, EMB,
                                                     xg, (size_t)GATES, nullptr, GATES, EMB / 32);
    {   // h0 = c0 = 0
        size_t th = (size_t)16 * HIDP * 2 / 4;
        zero_u32<<<blocks(th,256),256,0,stream>>>((unsigned*)hbf, th);
        size_t tc = (size_t)BATCH * HID;
        zero_u32<<<blocks(tc,256),256,0,stream>>>((unsigned*)cbuf, tc);
    }
    for (int t = 0; t < SEQ; ++t) {
        gemm_bf16_wmma<1,1><<<gridREC, wave, 0, stream>>>(hbf, HIDP, whh0_bf, HIDP,
                                                          gbuf, (size_t)GATES, nullptr, GATES, HIDP / 32);
        lstm_step<<<blocks((size_t)BATCH * HID, 256), 256, 0, stream>>>(
            gbuf, xg, b_ih0, b_hh0, cbuf, hbf, y0_bf, t);
    }

    // =================== layer 1 ===================
    gemm_bf16_wmma<2,4><<<gridXG, wave, 0, stream>>>(y0_bf, HIDP, wih1_bf, HIDP,
                                                     xg, (size_t)GATES, nullptr, GATES, HIDP / 32);
    {
        size_t th = (size_t)16 * HIDP * 2 / 4;
        zero_u32<<<blocks(th,256),256,0,stream>>>((unsigned*)hbf, th);
        size_t tc = (size_t)BATCH * HID;
        zero_u32<<<blocks(tc,256),256,0,stream>>>((unsigned*)cbuf, tc);
    }
    for (int t = 0; t < SEQ; ++t) {
        gemm_bf16_wmma<1,1><<<gridREC, wave, 0, stream>>>(hbf, HIDP, whh1_bf, HIDP,
                                                          gbuf, (size_t)GATES, nullptr, GATES, HIDP / 32);
        lstm_step<<<blocks((size_t)BATCH * HID, 256), 256, 0, stream>>>(
            gbuf, xg, b_ih1, b_hh1, cbuf, hbf, y1_bf, t);
    }

    // =================== output projection ===================
    gemm_bf16_wmma<2,4><<<gridLOG, wave, 0, stream>>>(y1_bf, HIDP, wout_bf, HIDP,
                                                      logits, (size_t)VOC, b_out, VOC, HIDP / 32);

    // =================== loss ===================
    zero_u32<<<1, 32, 0, stream>>>((unsigned*)accum, 2);
    loss_rows<<<BATCH * (SEQ - 1), 256, 0, stream>>>(logits, labels, accum);
    loss_finalize<<<1, 32, 0, stream>>>(accum, loss);
}
